// AttLayer_86423331930184
// MI455X (gfx1250) — matrix operand order repaired and checked by reading, not compiled, not run on hardware
//
#include <hip/hip_runtime.h>

// ---------------------------------------------------------------------------
// MI455X (gfx1250, wave32) attention layer.
// B=4, S=2048, D=A=1024, H=16, hd=64.  All GEMMs via v_wmma_f32_16x16x32_f16.
// Global->LDS staging via global_load_async_to_lds_b128 (ASYNCcnt) with
// double-buffered tiles; softmax in-register (S^T trick) with one shfl_xor.
// ---------------------------------------------------------------------------

#define LDT 80  // LDS row pitch in halfs: 64 data + 16 pad. 160B rows keep all
                // b128 accesses 16B-aligned; double-buffered tiles fit <64KB.

typedef __attribute__((ext_vector_type(16))) _Float16 v16h;
typedef __attribute__((ext_vector_type(8)))  _Float16 v8h;
typedef __attribute__((ext_vector_type(8)))  float    v8f;
typedef int vint4 __attribute__((vector_size(16)));  // builtin's pointee type

#if defined(__HIP_DEVICE_COMPILE__) && \
    __has_builtin(__builtin_amdgcn_global_load_async_to_lds_b128)
#define HAVE_ASYNC_LDS 1
#else
#define HAVE_ASYNC_LDS 0
#endif

// 16B global -> LDS direct copy (no VGPR round trip), tracked by ASYNCcnt.
__device__ __forceinline__ void async_b128(_Float16* lds, const _Float16* g) {
#if HAVE_ASYNC_LDS
  __builtin_amdgcn_global_load_async_to_lds_b128(
      (__attribute__((address_space(1))) vint4*)(unsigned long long)(const void*)g,
      (__attribute__((address_space(3))) vint4*)(unsigned int)(unsigned long long)(const void*)lds,
      0, 0);
#else
  (void)lds; (void)g;
#endif
}

template <int N>
__device__ __forceinline__ void wait_async() {
#if HAVE_ASYNC_LDS
#if __has_builtin(__builtin_amdgcn_s_wait_asynccnt)
  __builtin_amdgcn_s_wait_asynccnt(N);
#else
  asm volatile("s_wait_asynccnt %0" ::"i"(N) : "memory");
#endif
#endif
}

__device__ __forceinline__ v8f wmma16(v16h a, v16h b, v8f c) {
  // (neg_a, A, neg_b, B, c_mod, C, reuse_a, reuse_b)
  return __builtin_amdgcn_wmma_f32_16x16x32_f16(false, a, false, b, (short)0, c,
                                                false, false);
}

__device__ __forceinline__ v8f zero_v8f() {
  v8f z;
#pragma unroll
  for (int i = 0; i < 8; ++i) z[i] = 0.0f;
  return z;
}

// A fragment (16x32, M x K): lane r=lane&15 holds row M=r; elems 0..7 =
// K 8h+0..7, elems 8..15 = K 16+8h+0..7 (h = lane>>4). Row-major-in-K source.
__device__ __forceinline__ v16h frag_a(const _Float16* base, int ld) {
  const int lane = threadIdx.x & 31;
  const _Float16* p = base + (lane & 15) * ld + (lane >> 4) * 8;
  v8h lo = *(const v8h*)(p);
  v8h hi = *(const v8h*)(p + 16);
  v16h a;
#pragma unroll
  for (int i = 0; i < 8; ++i) { a[i] = lo[i]; a[i + 8] = hi[i]; }
  return a;
}

// B fragment (32x16, K x N): lane holds column N=lane&15; elems 0..7 =
// K 8h+0..7, elems 8..15 = K 16+8h+0..7.  Reads row N of a [N][K] (i.e. B-transposed row-major) buffer.
__device__ __forceinline__ v16h frag_b(const _Float16* base, int ld) {
  const int lane = threadIdx.x & 31;
  const _Float16* p = base + (lane & 15) * ld + (lane >> 4) * 8;
  v8h lo = *(const v8h*)(p);
  v8h hi = *(const v8h*)(p + 16);
  v16h b;
#pragma unroll
  for (int i = 0; i < 8; ++i) { b[i] = lo[i]; b[i + 8] = hi[i]; }
  return b;
}

// ------------------------------ converters --------------------------------

__global__ void cvt_f16_kernel(const float* __restrict__ src,
                               _Float16* __restrict__ dst, int n) {
  int i = blockIdx.x * blockDim.x + threadIdx.x;
  if (i < n) dst[i] = (_Float16)src[i];
}

// dst[n*K + k] = (f16) src[k*N + n]   (pre-transpose weights, K=rows of src)
__global__ void transpose_cvt_kernel(const float* __restrict__ src,
                                     _Float16* __restrict__ dst, int K, int N) {
  int i = blockIdx.x * blockDim.x + threadIdx.x;
  if (i < K * N) {
    int n = i / K;
    int k = i - n * K;
    dst[i] = (_Float16)src[(size_t)k * N + n];
  }
}

// ------------------------------ QKV GEMM ----------------------------------
// C[8192 x 3072] = Xh[8192 x 1024] * Wt^T (Wt stored [n][k]) + bias, f16 out.
__global__ __launch_bounds__(256) void qkv_gemm_kernel(
    const _Float16* __restrict__ Xh, const _Float16* __restrict__ Wt,
    const float* __restrict__ bq, const float* __restrict__ bk,
    const float* __restrict__ bv, _Float16* __restrict__ qkv) {
  __shared__ _Float16 As[2 * 128 * LDT];
  __shared__ _Float16 Bs[2 * 64 * LDT];
  const int tid = threadIdx.x;
  const int wid = tid >> 5;
  const int wm = wid >> 1;  // 0..3 -> 32-row strip
  const int wn = wid & 1;   // 0..1 -> 32-col strip
  const int m0 = blockIdx.x * 128;
  const int n0 = blockIdx.y * 64;

  auto stage = [&](int buf, int kb) {
    _Float16* Ad = As + buf * 128 * LDT;
    _Float16* Bd = Bs + buf * 64 * LDT;
#if HAVE_ASYNC_LDS
#pragma unroll
    for (int it = 0; it < 4; ++it) {  // 128 rows x 8 chunks of 8 halfs
      int c = tid + (it << 8), row = c >> 3, col = (c & 7) << 3;
      async_b128(&Ad[row * LDT + col],
                 &Xh[(size_t)(m0 + row) * 1024 + kb + col]);
    }
#pragma unroll
    for (int it = 0; it < 2; ++it) {  // 64 rows x 8 chunks
      int c = tid + (it << 8), row = c >> 3, col = (c & 7) << 3;
      async_b128(&Bd[row * LDT + col],
                 &Wt[(size_t)(n0 + row) * 1024 + kb + col]);
    }
#else
    v8h ta[4], tb[2];
#pragma unroll
    for (int it = 0; it < 4; ++it) {
      int c = tid + (it << 8);
      ta[it] = *(const v8h*)&Xh[(size_t)(m0 + (c >> 3)) * 1024 + kb +
                                ((c & 7) << 3)];
    }
#pragma unroll
    for (int it = 0; it < 2; ++it) {
      int c = tid + (it << 8);
      tb[it] = *(const v8h*)&Wt[(size_t)(n0 + (c >> 3)) * 1024 + kb +
                                ((c & 7) << 3)];
    }
#pragma unroll
    for (int it = 0; it < 4; ++it) {
      int c = tid + (it << 8);
      *(v8h*)&Ad[(c >> 3) * LDT + ((c & 7) << 3)] = ta[it];
    }
#pragma unroll
    for (int it = 0; it < 2; ++it) {
      int c = tid + (it << 8);
      *(v8h*)&Bd[(c >> 3) * LDT + ((c & 7) << 3)] = tb[it];
    }
#endif
  };

  v8f acc[2][2];
#pragma unroll
  for (int a = 0; a < 2; ++a)
#pragma unroll
    for (int b = 0; b < 2; ++b) acc[a][b] = zero_v8f();

  stage(0, 0);
  for (int i = 0; i < 16; ++i) {  // K = 1024, 64 per tile
    const int cur = i & 1;
    if (i + 1 < 16) stage(cur ^ 1, (i + 1) * 64);
#if HAVE_ASYNC_LDS
    if (i + 1 < 16) wait_async<6>(); else wait_async<0>();
#endif
    __syncthreads();
    const _Float16* Ac = As + cur * 128 * LDT;
    const _Float16* Bc = Bs + cur * 64 * LDT;
#pragma unroll
    for (int kk = 0; kk < 64; kk += 32) {
      v16h a0 = frag_a(&Ac[(wm * 32 + 0)  * LDT + kk], LDT);
      v16h a1 = frag_a(&Ac[(wm * 32 + 16) * LDT + kk], LDT);
      v16h b0 = frag_b(&Bc[(wn * 32 + 0)  * LDT + kk], LDT);
      v16h b1 = frag_b(&Bc[(wn * 32 + 16) * LDT + kk], LDT);
      acc[0][0] = wmma16(a0, b0, acc[0][0]);
      acc[0][1] = wmma16(a0, b1, acc[0][1]);
      acc[1][0] = wmma16(a1, b0, acc[1][0]);
      acc[1][1] = wmma16(a1, b1, acc[1][1]);
    }
    __syncthreads();
  }

  const int lane = tid & 31;
  const int r = lane & 15, hh = lane >> 4;
  const int which = n0 >> 10;  // uniform per block (64 | 1024)
  const float* bias = (which == 0) ? bq : (which == 1) ? bk : bv;
  _Float16* dst = qkv + (size_t)which * 8192 * 1024;
#pragma unroll
  for (int tm = 0; tm < 2; ++tm)
#pragma unroll
    for (int tn = 0; tn < 2; ++tn) {
      int mrow = m0 + wm * 32 + tm * 16 + 8 * hh;
      int acol = (n0 + wn * 32 + tn * 16 + r) & 1023;
      float bb = bias[acol];
#pragma unroll
      for (int i = 0; i < 8; ++i)
        dst[(size_t)(mrow + i) * 1024 + acol] =
            (_Float16)(acc[tm][tn][i] + bb);
    }
}

// ------------------------- RoPE + layout pack ------------------------------
// Qh,Kh: [b][h][s][d] (Q pre-scaled by 1/sqrt(hd));  Vt: [b][h][d][s].
__global__ void rope_pack_kernel(const _Float16* __restrict__ qkv,
                                 _Float16* __restrict__ Qh,
                                 _Float16* __restrict__ Kh,
                                 _Float16* __restrict__ Vt) {
  int idx = blockIdx.x * blockDim.x + threadIdx.x;  // token*H*32 pairs
  int j = idx & 31;
  int h = (idx >> 5) & 15;
  int token = idx >> 9;  // 0..8191
  int b = token >> 11;
  int s = token & 2047;

  size_t src = (size_t)token * 1024 + h * 64 + 2 * j;
  const _Float16* Qr = qkv;
  const _Float16* Kr = qkv + (size_t)8192 * 1024;
  const _Float16* Vr = qkv + (size_t)2 * 8192 * 1024;

  float q0 = (float)Qr[src], q1 = (float)Qr[src + 1];
  float k0 = (float)Kr[src], k1 = (float)Kr[src + 1];
  float v0 = (float)Vr[src], v1 = (float)Vr[src + 1];

  // theta = 10000^(2j/64);  ln(10000) = 9.210340371976184
  float inv = __expf(-(float)(2 * j) * (9.210340371976184f / 64.0f));
  float ang = (float)s * inv;
  float c, sn;
  __sincosf(ang, &c, &sn);

  const float scale = 0.125f;  // 1/sqrt(64), folded into Q
  size_t bh = (size_t)b * 16 + h;
  size_t dstq = (bh * 2048 + s) * 64 + 2 * j;
  Qh[dstq]     = (_Float16)((q0 * c - q1 * sn) * scale);
  Qh[dstq + 1] = (_Float16)((q0 * sn + q1 * c) * scale);
  Kh[dstq]     = (_Float16)(k0 * c - k1 * sn);
  Kh[dstq + 1] = (_Float16)(k0 * sn + k1 * c);
  size_t dstv = (bh * 64 + 2 * j) * 2048 + s;
  Vt[dstv]        = (_Float16)v0;
  Vt[dstv + 2048] = (_Float16)v1;
}

// --------------------------- flash attention -------------------------------
// Block = (b,h, 128 q-rows); 8 waves, each owns 16 q-rows.
// S^T = K_tile(64x64) * Q_wave^T(64x16)  -> softmax along accumulator elems.
// O^T = V^T_tile(64x64) * P^T(64x16)     -> accumulate with online rescale.
__global__ __launch_bounds__(256) void attn_kernel(
    const _Float16* __restrict__ Qh, const _Float16* __restrict__ Kh,
    const _Float16* __restrict__ Vt, _Float16* __restrict__ Oh) {
  __shared__ _Float16 Ks[2 * 64 * LDT];
  __shared__ _Float16 Vts[2 * 64 * LDT];
  __shared__ _Float16 Ps[8 * 16 * LDT];

  const int tid = threadIdx.x;
  const int wid = tid >> 5;
  const int lane = tid & 31;
  const int r = lane & 15, hh = lane >> 4;
  const int bh = blockIdx.x;    // b*16 + h
  const int qblk = blockIdx.y;  // 128 q-rows per block

  const size_t head = (size_t)bh * 2048 * 64;
  const _Float16* kbase = Kh + head;  // [s][d]
  const _Float16* vbase = Vt + head;  // [d][s]
  const _Float16* qwave = Qh + head + (size_t)(qblk * 128 + wid * 16) * 64;

  auto stage = [&](int buf, int kb) {
    _Float16* Kd = Ks + buf * 64 * LDT;
    _Float16* Vd = Vts + buf * 64 * LDT;
#if HAVE_ASYNC_LDS
#pragma unroll
    for (int it = 0; it < 2; ++it) {  // K tile: 64 rows(s) x 64(d)
      int c = tid + (it << 8), row = c >> 3, col = (c & 7) << 3;
      async_b128(&Kd[row * LDT + col],
                 &kbase[(size_t)(kb + row) * 64 + col]);
    }
#pragma unroll
    for (int it = 0; it < 2; ++it) {  // V^T tile: 64 rows(d) x 64(s)
      int c = tid + (it << 8), row = c >> 3, col = (c & 7) << 3;
      async_b128(&Vd[row * LDT + col],
                 &vbase[(size_t)row * 2048 + kb + col]);
    }
#else
    v8h tk[2], tv[2];
#pragma unroll
    for (int it = 0; it < 2; ++it) {
      int c = tid + (it << 8);
      tk[it] = *(const v8h*)&kbase[(size_t)(kb + (c >> 3)) * 64 +
                                   ((c & 7) << 3)];
      tv[it] = *(const v8h*)&vbase[(size_t)(c >> 3) * 2048 + kb +
                                   ((c & 7) << 3)];
    }
#pragma unroll
    for (int it = 0; it < 2; ++it) {
      int c = tid + (it << 8);
      *(v8h*)&Kd[(c >> 3) * LDT + ((c & 7) << 3)] = tk[it];
      *(v8h*)&Vd[(c >> 3) * LDT + ((c & 7) << 3)] = tv[it];
    }
#endif
  };

  // Q^T B-fragments for this wave (held in registers all kernel).
  v16h qb0 = frag_b(qwave + 0, 64);   // inner d = 0..31
  v16h qb1 = frag_b(qwave + 32, 64);  // inner d = 32..63

  float m_run = -1e30f, l_run = 0.0f;
  v8f oacc[4];
#pragma unroll
  for (int t = 0; t < 4; ++t) oacc[t] = zero_v8f();

  _Float16* pw = &Ps[wid * 16 * LDT];  // wave-private P^T staging

  stage(0, 0);
  for (int i = 0; i < 32; ++i) {  // S = 2048, 64 keys per tile
    const int cur = i & 1;
    if (i + 1 < 32) stage(cur ^ 1, (i + 1) * 64);
#if HAVE_ASYNC_LDS
    if (i + 1 < 32) wait_async<4>(); else wait_async<0>();
#endif
    __syncthreads();
    const _Float16* Kc = Ks + cur * 64 * LDT;
    const _Float16* Vc = Vts + cur * 64 * LDT;

    // scores^T : 4 M-tiles over the 64 keys, inner dim d=64 (2 wmma each)
    v8f st[4];
#pragma unroll
    for (int t = 0; t < 4; ++t) {
      v8f c = zero_v8f();
      c = wmma16(frag_a(&Kc[(t * 16) * LDT + 0], LDT), qb0, c);
      c = wmma16(frag_a(&Kc[(t * 16) * LDT + 32], LDT), qb1, c);
      st[t] = c;
    }

    // online softmax: this lane holds 32 of 64 key-scores for q-row r;
    // the partner (lane ^ 16) holds the other 32.
    float mloc = -3e38f;
#pragma unroll
    for (int t = 0; t < 4; ++t)
#pragma unroll
      for (int q = 0; q < 8; ++q) mloc = fmaxf(mloc, st[t][q]);
    mloc = fmaxf(mloc, __shfl_xor(mloc, 16));
    float mnew = fmaxf(m_run, mloc);
    float alpha = __expf(m_run - mnew);
    float ssum = 0.0f;
#pragma unroll
    for (int t = 0; t < 4; ++t)
#pragma unroll
      for (int q = 0; q < 8; ++q) {
        float p = __expf(st[t][q] - mnew);
        st[t][q] = p;
        ssum += p;
      }
    ssum += __shfl_xor(ssum, 16);
    l_run = l_run * alpha + ssum;
    m_run = mnew;
#pragma unroll
    for (int t = 0; t < 4; ++t) oacc[t] = oacc[t] * alpha;

    // stage P^T in LDS: row r (q), cols k = t*16 + 8*hh + i
#pragma unroll
    for (int t = 0; t < 4; ++t) {
      v8h ph;
#pragma unroll
      for (int q = 0; q < 8; ++q) ph[q] = (_Float16)st[t][q];
      *(v8h*)&pw[r * LDT + t * 16 + 8 * hh] = ph;
    }
    asm volatile("s_wait_dscnt 0" ::: "memory");  // wave-local LDS RAW fence

    v16h pb0 = frag_b(pw + 0, LDT);   // inner k = 0..31
    v16h pb1 = frag_b(pw + 32, LDT);  // inner k = 32..63
#pragma unroll
    for (int t = 0; t < 4; ++t) {
      oacc[t] = wmma16(frag_a(&Vc[(t * 16) * LDT + 0], LDT), pb0, oacc[t]);
      oacc[t] = wmma16(frag_a(&Vc[(t * 16) * LDT + 32], LDT), pb1, oacc[t]);
    }
    __syncthreads();
  }

  // write O: token = qblk*128 + wid*16 + r ; d = t*16 + 8*hh + i
  const float linv = 1.0f / l_run;
  const int b = bh >> 4, h = bh & 15;
  size_t token = (size_t)b * 2048 + qblk * 128 + wid * 16 + r;
  _Float16* op = Oh + token * 1024 + h * 64;
#pragma unroll
  for (int t = 0; t < 4; ++t) {
    v8h oh8;
#pragma unroll
    for (int q = 0; q < 8; ++q) oh8[q] = (_Float16)(oacc[t][q] * linv);
    *(v8h*)&op[t * 16 + 8 * hh] = oh8;
  }
}

// --------------------------- output projection -----------------------------
__global__ __launch_bounds__(256) void oproj_gemm_kernel(
    const _Float16* __restrict__ Oh, const _Float16* __restrict__ Wot,
    const float* __restrict__ bo, float* __restrict__ out) {
  __shared__ _Float16 As[2 * 128 * LDT];
  __shared__ _Float16 Bs[2 * 64 * LDT];
  const int tid = threadIdx.x;
  const int wid = tid >> 5;
  const int wm = wid >> 1, wn = wid & 1;
  const int m0 = blockIdx.x * 128;
  const int n0 = blockIdx.y * 64;

  auto stage = [&](int buf, int kb) {
    _Float16* Ad = As + buf * 128 * LDT;
    _Float16* Bd = Bs + buf * 64 * LDT;
#if HAVE_ASYNC_LDS
#pragma unroll
    for (int it = 0; it < 4; ++it) {
      int c = tid + (it << 8), row = c >> 3, col = (c & 7) << 3;
      async_b128(&Ad[row * LDT + col],
                 &Oh[(size_t)(m0 + row) * 1024 + kb + col]);
    }
#pragma unroll
    for (int it = 0; it < 2; ++it) {
      int c = tid + (it << 8), row = c >> 3, col = (c & 7) << 3;
      async_b128(&Bd[row * LDT + col],
                 &Wot[(size_t)(n0 + row) * 1024 + kb + col]);
    }
#else
    v8h ta[4], tb[2];
#pragma unroll
    for (int it = 0; it < 4; ++it) {
      int c = tid + (it << 8);
      ta[it] = *(const v8h*)&Oh[(size_t)(m0 + (c >> 3)) * 1024 + kb +
                                ((c & 7) << 3)];
    }
#pragma unroll
    for (int it = 0; it < 2; ++it) {
      int c = tid + (it << 8);
      tb[it] = *(const v8h*)&Wot[(size_t)(n0 + (c >> 3)) * 1024 + kb +
                                 ((c & 7) << 3)];
    }
#pragma unroll
    for (int it = 0; it < 4; ++it) {
      int c = tid + (it << 8);
      *(v8h*)&Ad[(c >> 3) * LDT + ((c & 7) << 3)] = ta[it];
    }
#pragma unroll
    for (int it = 0; it < 2; ++it) {
      int c = tid + (it << 8);
      *(v8h*)&Bd[(c >> 3) * LDT + ((c & 7) << 3)] = tb[it];
    }
#endif
  };

  v8f acc[2][2];
#pragma unroll
  for (int a = 0; a < 2; ++a)
#pragma unroll
    for (int b = 0; b < 2; ++b) acc[a][b] = zero_v8f();

  stage(0, 0);
  for (int i = 0; i < 16; ++i) {
    const int cur = i & 1;
    if (i + 1 < 16) stage(cur ^ 1, (i + 1) * 64);
#if HAVE_ASYNC_LDS
    if (i + 1 < 16) wait_async<6>(); else wait_async<0>();
#endif
    __syncthreads();
    const _Float16* Ac = As + cur * 128 * LDT;
    const _Float16* Bc = Bs + cur * 64 * LDT;
#pragma unroll
    for (int kk = 0; kk < 64; kk += 32) {
      v16h a0 = frag_a(&Ac[(wm * 32 + 0)  * LDT + kk], LDT);
      v16h a1 = frag_a(&Ac[(wm * 32 + 16) * LDT + kk], LDT);
      v16h b0 = frag_b(&Bc[(wn * 32 + 0)  * LDT + kk], LDT);
      v16h b1 = frag_b(&Bc[(wn * 32 + 16) * LDT + kk], LDT);
      acc[0][0] = wmma16(a0, b0, acc[0][0]);
      acc[0][1] = wmma16(a0, b1, acc[0][1]);
      acc[1][0] = wmma16(a1, b0, acc[1][0]);
      acc[1][1] = wmma16(a1, b1, acc[1][1]);
    }
    __syncthreads();
  }

  const int lane = tid & 31;
  const int r = lane & 15, hh = lane >> 4;
#pragma unroll
  for (int tm = 0; tm < 2; ++tm)
#pragma unroll
    for (int tn = 0; tn < 2; ++tn) {
      int mrow = m0 + wm * 32 + tm * 16 + 8 * hh;
      int ncol = n0 + wn * 32 + tn * 16 + r;
      float bb = bo[ncol];
#pragma unroll
      for (int i = 0; i < 8; ++i)
        out[(size_t)(mrow + i) * 1024 + ncol] = acc[tm][tn][i] + bb;
    }
}

// ------------------------------- launcher ----------------------------------

extern "C" void kernel_launch(void* const* d_in, const int* in_sizes, int n_in,
                              void* d_out, int out_size, void* d_ws,
                              size_t ws_size, hipStream_t stream) {
  const float* x  = (const float*)d_in[0];
  const float* Wq = (const float*)d_in[1];
  const float* bq = (const float*)d_in[2];
  const float* Wk = (const float*)d_in[3];
  const float* bk = (const float*)d_in[4];
  const float* Wv = (const float*)d_in[5];
  const float* bv = (const float*)d_in[6];
  const float* Wo = (const float*)d_in[7];
  const float* bo = (const float*)d_in[8];
  float* out = (float*)d_out;

  char* w = (char*)d_ws;
  const size_t MB = 1024 * 1024;
  _Float16* Xh   = (_Float16*)(w + 0);         // 16 MB  (8192x1024 f16)
  _Float16* Wt   = (_Float16*)(w + 16 * MB);   //  6 MB  (3072x1024 f16, [n][k])
  _Float16* Wot  = (_Float16*)(w + 22 * MB);   //  2 MB  (1024x1024 f16, [n][k])
  _Float16* QKVr = (_Float16*)(w + 24 * MB);   // 48 MB  ([3][8192][1024] f16)
  _Float16* Oh   = (_Float16*)(w + 24 * MB);   // 16 MB  (overlays dead QKVr)
  _Float16* Qh   = (_Float16*)(w + 72 * MB);   // 16 MB  [b][h][s][d]
  _Float16* Kh   = (_Float16*)(w + 88 * MB);   // 16 MB  [b][h][s][d]
  _Float16* Vt   = (_Float16*)(w + 104 * MB);  // 16 MB  [b][h][d][s]
  (void)ws_size; (void)in_sizes; (void)n_in; (void)out_size;

  // 1) precision conversion + weight pre-transpose
  cvt_f16_kernel<<<(8192 * 1024 + 255) / 256, 256, 0, stream>>>(x, Xh,
                                                                8192 * 1024);
  const int WN = 1024 * 1024;
  transpose_cvt_kernel<<<(WN + 255) / 256, 256, 0, stream>>>(Wq, Wt, 1024, 1024);
  transpose_cvt_kernel<<<(WN + 255) / 256, 256, 0, stream>>>(Wk, Wt + WN, 1024, 1024);
  transpose_cvt_kernel<<<(WN + 255) / 256, 256, 0, stream>>>(Wv, Wt + 2 * WN, 1024, 1024);
  transpose_cvt_kernel<<<(WN + 255) / 256, 256, 0, stream>>>(Wo, Wot, 1024, 1024);

  // 2) fused QKV projection (WMMA, async double-buffered staging)
  qkv_gemm_kernel<<<dim3(64, 48), 256, 0, stream>>>(Xh, Wt, bq, bk, bv, QKVr);

  // 3) RoPE + attention-friendly packing (Q scaled, V transposed)
  rope_pack_kernel<<<(8192 * 16 * 32) / 256, 256, 0, stream>>>(QKVr, Qh, Kh, Vt);

  // 4) flash attention (WMMA, online softmax, async double-buffered K/V)
  attn_kernel<<<dim3(64, 16), 256, 0, stream>>>(Qh, Kh, Vt, Oh);

  // 5) output projection (WMMA, f32 out + bias)
  oproj_gemm_kernel<<<dim3(64, 16), 256, 0, stream>>>(Oh, Wot, bo, out);
}
